// FH_Mamba_5403068858474
// MI455X (gfx1250) — matrix operand order repaired and checked by reading, not compiled, not run on hardware
//
#include <hip/hip_runtime.h>
#include <stdint.h>

typedef __attribute__((ext_vector_type(16))) _Float16 v16h;
typedef __attribute__((ext_vector_type(8)))  float    v8f;

#define EPSN 1e-5f

__device__ __forceinline__ float actf(float x, int a){
  if (a == 1) return x / (1.f + __expf(-x));                       // silu
  if (a == 2) return 0.5f * x * (1.f + erff(x * 0.70710678118f));  // gelu (exact)
  if (a == 3) return (x > 20.f) ? x : log1pf(__expf(x));           // softplus
  if (a == 4) return 1.f / (1.f + __expf(-x));                     // sigmoid
  return x;
}

// ---------------------------------------------------------------------------
// Implicit-GEMM conv. Block = 256 threads = 8 waves, macro-tile BM=32 x BN=256.
// Each wave: 2 M-subtiles x 2 N-subtiles -> 4 x v_wmma_f32_16x16x32_f16 per
// K-step with full A/B fragment reuse. B is gathered (im2col) with f32->f16
// conversion directly into per-lane fragment layout (2x ds_load_b128 to read).
// A (weight) tile streams via global_load_async_to_lds_b32 when in-bounds.
// KHW is the compile-time kernel footprint (1 or 9).
// ---------------------------------------------------------------------------
template <int KHW>
__global__ __launch_bounds__(256)
void conv_wmma(const float* __restrict__ in, const float* __restrict__ w,
               const float* __restrict__ bias, float* __restrict__ out,
               int Cin, int H, int W, int Ho, int woShift,
               int stride, int pad, int Cout,
               long ibs, long obs, int coff)
{
  __shared__ float As[32][32];                       // weights tile (f32)
  __shared__ __align__(32) _Float16 Bf[16][32][16];  // B fragments (f16, per-lane 32B)
  const int tid  = threadIdx.x;
  const int lane = tid & 31;
  const int wave = tid >> 5;
  const int Wo   = 1 << woShift;
  const int Npix = Ho << woShift;
  const int n0 = blockIdx.x * 256;
  const int m0 = blockIdx.y * 32;
  const int z  = blockIdx.z;
  const int Ktot = Cin * KHW;
  const int ksteps = (Ktot + 31) >> 5;
  const float* inz = in + (long)z * ibs;

  v8f acc00 = {0,0,0,0,0,0,0,0}, acc01 = acc00, acc10 = acc00, acc11 = acc00;

  const int row  = lane & 15;
  const int half = lane >> 4;
  const int ntb  = wave * 2;

  for (int ks = 0; ks < ksteps; ++ks) {
    const int k0 = ks << 5;
    // ---- A tile: 32 rows x 32 k (f32) ----
    const bool fullA = (m0 + 32 <= Cout) && (k0 + 32 <= Ktot);
    if (fullA) {
      for (int t = tid; t < 1024; t += 256) {
        int r = t >> 5, k = t & 31;
        unsigned lds = (unsigned)(unsigned long long)(void*)&As[r][k]; // low 32b = LDS offset
        unsigned long long g = (unsigned long long)(const void*)(w + (long)(m0 + r) * Ktot + k0 + k);
        asm volatile("global_load_async_to_lds_b32 %0, %1, off" :: "v"(lds), "v"(g) : "memory");
      }
      asm volatile("s_wait_asynccnt 0" ::: "memory");
    } else {
      for (int t = tid; t < 1024; t += 256) {
        int r = t >> 5, k = t & 31;
        float v = 0.f;
        if (m0 + r < Cout && k0 + k < Ktot) v = w[(long)(m0 + r) * Ktot + k0 + k];
        As[r][k] = v;
      }
    }
    // ---- B fragments: 16 N-subtiles x 32 lanes x 16 values (im2col gather) ----
    for (int s = tid; s < 8192; s += 256) {
      int nt = s >> 9, ln = (s >> 4) & 31, j = s & 15;
      int k  = ((ln >> 4) << 4) + j;          // K within 32-chunk
      int p  = n0 + nt * 16 + (ln & 15);      // output pixel
      int kk = k0 + k;
      float v = 0.f;
      if (kk < Ktot && p < Npix) {
        int ci, kh, kw;
        if (KHW == 9) { ci = kk / 9; int rem = kk - ci * 9; kh = rem / 3; kw = rem - kh * 3; }
        else          { ci = kk; kh = 0; kw = 0; }
        int oy = p >> woShift, ox = p & (Wo - 1);
        int iy = oy * stride - pad + kh;
        int ix = ox * stride - pad + kw;
        if ((unsigned)iy < (unsigned)H && (unsigned)ix < (unsigned)W)
          v = inz[(long)ci * H * W + (long)iy * W + ix];
      }
      Bf[nt][ln][j] = (_Float16)v;
    }
    __syncthreads();
    // ---- A fragments (contiguous k = 16*half + j, matching B) ----
    v16h af0, af1;
    #pragma unroll
    for (int j = 0; j < 16; ++j) {
      int r = j >> 1, lo = j & 1;
      int k = half * 16 + 2 * r + lo;
      af0[j] = (_Float16)As[row][k];
      af1[j] = (_Float16)As[16 + row][k];
    }
    // ---- B fragments: contiguous per-lane 32B reads ----
    v16h bf0 = *(const v16h*)&Bf[ntb][lane][0];
    v16h bf1 = *(const v16h*)&Bf[ntb + 1][lane][0];
    acc00 = __builtin_amdgcn_wmma_f32_16x16x32_f16(false, af0, false, bf0, (short)0, acc00, false, false);
    acc01 = __builtin_amdgcn_wmma_f32_16x16x32_f16(false, af0, false, bf1, (short)0, acc01, false, false);
    acc10 = __builtin_amdgcn_wmma_f32_16x16x32_f16(false, af1, false, bf0, (short)0, acc10, false, false);
    acc11 = __builtin_amdgcn_wmma_f32_16x16x32_f16(false, af1, false, bf1, (short)0, acc11, false, false);
    __syncthreads();
  }
  // ---- epilogue: VGPR r -> M = r + 8*half, N = lane%16 ----
  const int col0 = n0 + ntb * 16 + row;
  const int col1 = col0 + 16;
  #pragma unroll
  for (int r = 0; r < 8; ++r) {
    int ma = m0 + half * 8 + r;        // mi = 0
    int mb = ma + 16;                  // mi = 1
    if (ma < Cout) {
      float bv = bias ? bias[ma] : 0.f;
      if (col0 < Npix) out[(long)z * obs + (long)(coff + ma) * Npix + col0] = acc00[r] + bv;
      if (col1 < Npix) out[(long)z * obs + (long)(coff + ma) * Npix + col1] = acc01[r] + bv;
    }
    if (mb < Cout) {
      float bv = bias ? bias[mb] : 0.f;
      if (col0 < Npix) out[(long)z * obs + (long)(coff + mb) * Npix + col0] = acc10[r] + bv;
      if (col1 < Npix) out[(long)z * obs + (long)(coff + mb) * Npix + col1] = acc11[r] + bv;
    }
  }
}

// ---------------------------------------------------------------------------
__global__ void dwconv3_k(const float* __restrict__ in, const float* __restrict__ w,
                          const float* __restrict__ b, float* __restrict__ out,
                          int C, int H, int W, long ibs, long obs, int coff, long total)
{
  long i = (long)blockIdx.x * blockDim.x + threadIdx.x;
  if (i >= total) return;
  int HW = H * W;
  int p = (int)(i % HW); long rc = i / HW;
  int c = (int)(rc % C); int z = (int)(rc / C);
  int y = p / W, x = p % W;
  const float* ip = in + (long)z * ibs + (long)c * HW;
  float s = b[c];
  #pragma unroll
  for (int dy = -1; dy <= 1; ++dy)
    #pragma unroll
    for (int dx = -1; dx <= 1; ++dx) {
      int yy = y + dy, xx = x + dx;
      if ((unsigned)yy < (unsigned)H && (unsigned)xx < (unsigned)W)
        s += w[c * 9 + (dy + 1) * 3 + (dx + 1)] * ip[yy * W + xx];
    }
  out[(long)z * obs + (long)(coff + c) * HW + p] = s;
}

__global__ __launch_bounds__(256) void inorm_k(float* x, int HW, int act)
{
  long base = (long)blockIdx.x * HW;
  float s = 0.f, s2 = 0.f;
  for (int i = threadIdx.x; i < HW; i += 256) { float v = x[base + i]; s += v; s2 += v * v; }
  __shared__ float sh[512];
  sh[threadIdx.x] = s; sh[256 + threadIdx.x] = s2; __syncthreads();
  for (int o = 128; o > 0; o >>= 1) {
    if (threadIdx.x < o) { sh[threadIdx.x] += sh[threadIdx.x + o]; sh[256 + threadIdx.x] += sh[256 + threadIdx.x + o]; }
    __syncthreads();
  }
  float mu = sh[0] / HW, var = sh[256] / HW - mu * mu;
  float rs = rsqrtf(var + EPSN);
  for (int i = threadIdx.x; i < HW; i += 256)
    x[base + i] = actf((x[base + i] - mu) * rs, act);
}

__global__ __launch_bounds__(256) void bnorm_gelu_k(float* x, const float* g, const float* bt,
                                                    int C, int N, int HW)
{
  int c = blockIdx.x;
  long cs = (long)c * HW, ns = (long)C * HW;
  int tot = N * HW;
  float s = 0.f, s2 = 0.f;
  for (int i = threadIdx.x; i < tot; i += 256) {
    int n = i / HW, p = i % HW;
    float v = x[(long)n * ns + cs + p]; s += v; s2 += v * v;
  }
  __shared__ float sh[512];
  sh[threadIdx.x] = s; sh[256 + threadIdx.x] = s2; __syncthreads();
  for (int o = 128; o > 0; o >>= 1) {
    if (threadIdx.x < o) { sh[threadIdx.x] += sh[threadIdx.x + o]; sh[256 + threadIdx.x] += sh[256 + threadIdx.x + o]; }
    __syncthreads();
  }
  float mu = sh[0] / tot, var = sh[256] / tot - mu * mu;
  float sc = g[c] * rsqrtf(var + EPSN);
  float sf = bt[c] - mu * sc;
  for (int i = threadIdx.x; i < tot; i += 256) {
    int n = i / HW, p = i % HW;
    long a = (long)n * ns + cs + p;
    x[a] = actf(x[a] * sc + sf, 2);
  }
}

__global__ void lnorm_k(const float* X, const float* g, const float* b, float* Y, long rows, int C)
{
  long r = (long)blockIdx.x * blockDim.x + threadIdx.x;
  if (r >= rows) return;
  const float* xp = X + r * C;
  float mu = 0.f;
  for (int i = 0; i < C; ++i) mu += xp[i];
  mu /= C;
  float var = 0.f;
  for (int i = 0; i < C; ++i) { float d = xp[i] - mu; var += d * d; }
  float rs = rsqrtf(var / C + EPSN);
  float* yp = Y + r * C;
  for (int i = 0; i < C; ++i) yp[i] = (xp[i] - mu) * rs * g[i] + b[i];
}

__global__ void lin_k(const float* X, int ldx, int xo, const float* Wm, const float* bias,
                      const float* resid, int ldr, float* Y, int ldy,
                      long rows, int K, int N, int act)
{
  long i = (long)blockIdx.x * blockDim.x + threadIdx.x;
  if (i >= rows * N) return;
  long r = i / N; int o = (int)(i % N);
  float s = bias ? bias[o] : 0.f;
  const float* xp = X + r * ldx + xo;
  for (int k = 0; k < K; ++k) s += xp[k] * Wm[k * N + o];
  if (resid) s += resid[r * ldr + o];
  Y[r * ldy + o] = actf(s, act);
}

__global__ void mconv1d_k(const float* xz, const float* w, const float* b, float* xs2, int L)
{
  long i = (long)blockIdx.x * blockDim.x + threadIdx.x;
  if (i >= 4L * L * 16) return;
  int d = (int)(i & 15); long rl = i >> 4;
  int l = (int)(rl % L); int bb = (int)(rl / L);
  float s = b[d];
  #pragma unroll
  for (int t = 0; t < 4; ++t) {
    int ls = l - 3 + t;
    if (ls >= 0) s += w[d * 4 + t] * xz[((long)bb * L + ls) * 32 + d];
  }
  xs2[i] = s / (1.f + __expf(-s));
}

__global__ void negexp_k(const float* alog, float* A)
{
  int i = threadIdx.x;
  if (i < 256) A[i] = -__expf(alog[i]);
}

__global__ __launch_bounds__(256)
void scan_k(const float* __restrict__ dt, const float* __restrict__ dbl,
            const float* __restrict__ xs2, const float* __restrict__ A,
            float* __restrict__ ys, int L)
{
  int b = blockIdx.x;
  int d = threadIdx.x >> 4, n = threadIdx.x & 15;
  float a = A[d * 16 + n];
  float h = 0.f;
  for (int l = 0; l < L; ++l) {
    long r = (long)b * L + l;
    __builtin_prefetch(&dt[(r + 8) * 16], 0, 0);
    float dtv = dt[r * 16 + d];
    float Bv  = dbl[r * 33 + 1 + n];
    float Cv  = dbl[r * 33 + 17 + n];
    float xv  = xs2[r * 16 + d];
    h = __expf(dtv * a) * h + dtv * Bv * xv;
    float p = h * Cv;
    p += __shfl_xor(p, 1, 16);
    p += __shfl_xor(p, 2, 16);
    p += __shfl_xor(p, 4, 16);
    p += __shfl_xor(p, 8, 16);
    if (n == 0) ys[r * 16 + d] = p;
  }
}

__global__ void mgate_k(const float* ys, const float* xs2, const float* Dp, const float* xz, float* yg)
{
  long i = (long)blockIdx.x * blockDim.x + threadIdx.x;
  if (i >= 262144) return;
  long r = i >> 4; int d = (int)(i & 15);
  float y = ys[i] + xs2[i] * Dp[d];
  float z = xz[r * 32 + 16 + d];
  yg[i] = y * (z / (1.f + __expf(-z)));
}

// ---------------- Hilbert tables ----------------
__global__ void hil_k(int* hil)
{
  int dd = blockIdx.x * blockDim.x + threadIdx.x;
  if (dd >= 1024) return;
  int t = dd, x = 0, y = 0;
  for (int s = 1; s < 32; s <<= 1) {
    int rx = 1 & (t >> 1);
    int ry = 1 & (t ^ rx);
    if (ry == 0) {
      if (rx == 1) { x = s - 1 - x; y = s - 1 - y; }
      int tmp = x; x = y; y = tmp;
    }
    x += s * rx; y += s * ry; t >>= 2;
  }
  hil[dd] = x * 32 + y;
}
__global__ void invhil_k(const int* hil, int* inv)
{
  int j = blockIdx.x * blockDim.x + threadIdx.x;
  if (j < 1024) inv[hil[j]] = j;
}
__global__ void hgather_k(const float* feat, const int* hil, float* seq)
{
  long i = (long)blockIdx.x * blockDim.x + threadIdx.x;
  if (i >= 262144) return;
  int c = (int)(i & 15); int j = (int)((i >> 4) & 1023);
  int t = (int)((i >> 14) & 3); int b = (int)(i >> 16);
  seq[i] = feat[(((long)(b * 4 + t) * 16 + c) << 10) + hil[j]];
}
__global__ void hscatter_k(const float* seqf, const int* inv, float* cat)
{
  long i = (long)blockIdx.x * blockDim.x + threadIdx.x;
  if (i >= 262144) return;
  int p = (int)(i & 1023); int c = (int)((i >> 10) & 15);
  int t = (int)((i >> 14) & 3); int b = (int)(i >> 16);
  cat[(long)b * 131072 + (t * 16 + c) * 1024 + p] =
      seqf[(((long)(b * 4 + t) * 1024 + inv[p]) << 4) + c];
}

// ---------------- Haar ----------------
__global__ void haarf_k(const float* x, float* y)
{
  long i = (long)blockIdx.x * blockDim.x + threadIdx.x;
  if (i >= 4L * 64 * 256) return;
  int p = (int)(i & 255); int c = (int)((i >> 8) & 63); int z = (int)(i >> 14);
  int yy = p >> 4, xx = p & 15;
  long base = (long)z * 65536 + (long)c * 1024;
  float x1 = x[base + (2 * yy) * 32 + 2 * xx];
  float x2 = x[base + (2 * yy + 1) * 32 + 2 * xx];
  float x3 = x[base + (2 * yy) * 32 + 2 * xx + 1];
  float x4 = x[base + (2 * yy + 1) * 32 + 2 * xx + 1];
  long ob = (long)z * 65536;
  y[ob + (long)c * 256 + p]          = ( x1 + x2 + x3 + x4) * 0.5f;
  y[ob + (long)(64 + c) * 256 + p]   = (-x1 + x2 - x3 + x4) * 0.5f;
  y[ob + (long)(128 + c) * 256 + p]  = (-x1 - x2 + x3 + x4) * 0.5f;
  y[ob + (long)(192 + c) * 256 + p]  = ( x1 - x2 - x3 + x4) * 0.5f;
}
__global__ void haari_k(const float* yc, float* out, long obs, int coff)
{
  long i = (long)blockIdx.x * blockDim.x + threadIdx.x;
  if (i >= 4L * 64 * 256) return;
  int p = (int)(i & 255); int c = (int)((i >> 8) & 63); int z = (int)(i >> 14);
  long zb = (long)z * 65536;
  float ll = yc[zb + (long)c * 256 + p];
  float lh = yc[zb + (long)(64 + c) * 256 + p];
  float hl = yc[zb + (long)(128 + c) * 256 + p];
  float hh = yc[zb + (long)(192 + c) * 256 + p];
  float x1 = (ll - lh - hl + hh) * 0.5f;
  float x2 = (ll + lh - hl - hh) * 0.5f;
  float x3 = (ll - lh + hl - hh) * 0.5f;
  float x4 = (ll + lh + hl + hh) * 0.5f;
  int yy = p >> 4, xx = p & 15;
  long ob = (long)z * obs + (long)(coff + c) * 1024;
  out[ob + (2 * yy) * 32 + 2 * xx] = x1;
  out[ob + (2 * yy + 1) * 32 + 2 * xx] = x2;
  out[ob + (2 * yy) * 32 + 2 * xx + 1] = x3;
  out[ob + (2 * yy + 1) * 32 + 2 * xx + 1] = x4;
}

__global__ void upbil_k(const float* in, float* out)
{
  long i = (long)blockIdx.x * blockDim.x + threadIdx.x;
  if (i >= 262144) return;
  int p = (int)(i & 1023); int pl = (int)(i >> 10);
  int oy = p >> 5, ox = p & 31;
  float sy = oy * 0.5f - 0.25f, sx = ox * 0.5f - 0.25f;
  int y0 = (int)floorf(sy), x0 = (int)floorf(sx);
  float fy = sy - y0, fx = sx - x0;
  int y0c = min(max(y0, 0), 15), y1c = min(max(y0 + 1, 0), 15);
  int x0c = min(max(x0, 0), 15), x1c = min(max(x0 + 1, 0), 15);
  const float* ip = in + (long)pl * 256;
  float v00 = ip[y0c * 16 + x0c], v01 = ip[y0c * 16 + x1c];
  float v10 = ip[y1c * 16 + x0c], v11 = ip[y1c * 16 + x1c];
  out[i] = (1 - fy) * ((1 - fx) * v00 + fx * v01) + fy * ((1 - fx) * v10 + fx * v11);
}

__global__ void pshuf_k(const float* in, float* out, int Bn, int C4, int Hh, int Ww)
{
  int C = C4 >> 2;
  long total = (long)Bn * C4 * Hh * Ww;
  long i = (long)blockIdx.x * blockDim.x + threadIdx.x;
  if (i >= total) return;
  int W2 = 2 * Ww, H2 = 2 * Hh;
  int ox = (int)(i % W2); long t = i / W2;
  int oy = (int)(t % H2); t /= H2;
  int c = (int)(t % C); int z = (int)(t / C);
  int r1 = oy & 1, r2 = ox & 1;
  out[i] = in[((long)z * C4 + c * 4 + r1 * 2 + r2) * Hh * Ww + (long)(oy >> 1) * Ww + (ox >> 1)];
}

__global__ void add_k(const float* a, const float* b, float* o, long n)
{
  long i = (long)blockIdx.x * blockDim.x + threadIdx.x;
  if (i < n) o[i] = a[i] + b[i];
}

__global__ void blend_k(const float* g, const float* cat, float* bl)
{
  long i = (long)blockIdx.x * blockDim.x + threadIdx.x;
  if (i >= 262144) return;
  int p = (int)(i & 1023); int tc = (int)((i >> 10) & 63); int b = (int)(i >> 16);
  float s = 1.f / (1.f + __expf(-g[i]));
  float m = cat[(long)b * 131072 + tc * 1024 + p];
  float w = cat[(long)b * 131072 + 65536 + tc * 1024 + p];
  bl[i] = s * m + (1.f - s) * w;
}

__global__ void ri_k(const float* emb, const float* w, const float* b, float* z)
{
  long i = (long)blockIdx.x * blockDim.x + threadIdx.x;
  if (i >= 262144) return;
  int hw = (int)(i & 1023); int t = (int)((i >> 10) & 3);
  int o = (int)((i >> 12) & 15); int bb = (int)(i >> 16);
  float s = b[o];
  for (int c = 0; c < 64; ++c)
    s += w[o * 64 + c] * emb[(((long)(bb * 64 + c) * 4 + t) << 10) + hw];
  z[i] = s;
}
__global__ void ro_k(const float* zin, const float* w, const float* b, float* out)
{
  long i = (long)blockIdx.x * blockDim.x + threadIdx.x;
  if (i >= 1048576) return;
  int hw = (int)(i & 1023); int t = (int)((i >> 10) & 3);
  int o = (int)((i >> 12) & 63); int bb = (int)(i >> 18);
  float s = b[o];
  for (int c = 0; c < 16; ++c)
    s += w[o * 16 + c] * zin[(((long)bb * 64 + c * 4 + t) << 10) + hw];
  out[i] = s;
}

__global__ __launch_bounds__(256)
void finalize_k(const float* pred, const float* tgt, float* out, float* loss, long n)
{
  float acc = 0.f;
  for (long i = (long)blockIdx.x * blockDim.x + threadIdx.x; i < n; i += (long)gridDim.x * blockDim.x) {
    float p = fminf(fmaxf(pred[i], 0.f), 1.f);
    out[i] = p;
    float e = p - tgt[i], ae = fabsf(e);
    acc += (ae < 1.f) ? 0.5f * e * e : ae - 0.5f;
  }
  __shared__ float sh[256];
  sh[threadIdx.x] = acc; __syncthreads();
  for (int o = 128; o > 0; o >>= 1) {
    if (threadIdx.x < o) sh[threadIdx.x] += sh[threadIdx.x + o];
    __syncthreads();
  }
  if (threadIdx.x == 0) atomicAdd(loss, sh[0] / (float)n);
}

// ===========================================================================
static inline void launch_conv(hipStream_t s, const float* in, const float* w, const float* bias,
                               float* out, int batch, int Cin, int H, int W,
                               int KH, int KW, int stride, int pad, int Cout,
                               long ibs, long obs, int coff)
{
  int Ho = (H + 2 * pad - KH) / stride + 1;
  int Wo = (W + 2 * pad - KW) / stride + 1;
  int woShift = __builtin_ctz(Wo);                 // all Wo here are powers of two
  dim3 g((Ho * Wo + 255) / 256, (Cout + 31) / 32, batch);
  if (KH == 3)
    conv_wmma<9><<<g, 256, 0, s>>>(in, w, bias, out, Cin, H, W, Ho, woShift, stride, pad, Cout, ibs, obs, coff);
  else
    conv_wmma<1><<<g, 256, 0, s>>>(in, w, bias, out, Cin, H, W, Ho, woShift, stride, pad, Cout, ibs, obs, coff);
}

extern "C" void kernel_launch(void* const* d_in, const int* in_sizes, int n_in,
                              void* d_out, int out_size, void* d_ws, size_t ws_size,
                              hipStream_t stream)
{
  (void)in_sizes; (void)n_in; (void)out_size; (void)ws_size;
  const float* IN[200];
  for (int i = 0; i < n_in && i < 200; ++i) IN[i] = (const float*)d_in[i];
  int pi = 0;
  const float* input_x = IN[pi++];
  const float* targets = IN[pi++];
  struct DBp { const float *b1w,*b1b,*b2w,*b2b,*dww,*dwb,*fw,*fb; } enc[4];
  for (int i = 0; i < 4; ++i) {
    enc[i].b1w = IN[pi++]; enc[i].b1b = IN[pi++]; enc[i].b2w = IN[pi++]; enc[i].b2b = IN[pi++];
    enc[i].dww = IN[pi++]; enc[i].dwb = IN[pi++]; enc[i].fw  = IN[pi++]; enc[i].fb  = IN[pi++];
  }
  struct DLp { const float *w,*b; } dec[5];
  for (int i = 0; i < 5; ++i) { dec[i].w = IN[pi++]; dec[i].b = IN[pi++]; }
  const float* ri_w = IN[pi++]; const float* ri_b = IN[pi++];
  const float* ro_w = IN[pi++]; const float* ro_b = IN[pi++];
  struct MBp {
    const float *ln_g,*ln_b,*in_w,*in_b,*conv_w,*conv_b,*xp_w,*dt_w,*dt_b,*A_log,*D,*out_w,*out_b,
                *ln2_g,*ln2_b,*w1,*b1,*w2,*b2,
                *fq_w,*fq_b,*fq_g,*fq_bt,*dw_w,*dw_b,*pw_w,*pw_b,*bn_g,*bn_b,*g_w,*g_b,*o_w,*o_b;
  } hid[3];
  for (int i = 0; i < 3; ++i) {
    MBp& m = hid[i];
    m.ln_g=IN[pi++]; m.ln_b=IN[pi++]; m.in_w=IN[pi++]; m.in_b=IN[pi++];
    m.conv_w=IN[pi++]; m.conv_b=IN[pi++]; m.xp_w=IN[pi++]; m.dt_w=IN[pi++]; m.dt_b=IN[pi++];
    m.A_log=IN[pi++]; m.D=IN[pi++]; m.out_w=IN[pi++]; m.out_b=IN[pi++];
    m.ln2_g=IN[pi++]; m.ln2_b=IN[pi++]; m.w1=IN[pi++]; m.b1=IN[pi++]; m.w2=IN[pi++]; m.b2=IN[pi++];
    m.fq_w=IN[pi++]; m.fq_b=IN[pi++]; m.fq_g=IN[pi++]; m.fq_bt=IN[pi++];
    m.dw_w=IN[pi++]; m.dw_b=IN[pi++]; m.pw_w=IN[pi++]; m.pw_b=IN[pi++];
    m.bn_g=IN[pi++]; m.bn_b=IN[pi++]; m.g_w=IN[pi++]; m.g_b=IN[pi++]; m.o_w=IN[pi++]; m.o_b=IN[pi++];
  }

  // ---- workspace layout (floats) ----
  float* W0 = (float*)d_ws;               // 33,554,432 : concat @128^2, decoder scratch
  float* W1 = W0 + 33554432;              // 16,777,216
  float* W2 = W1 + 16777216;              // 16,777,216
  float* W3 = W2 + 16777216;              // 16,777,216 : enc1 (kept)
  float* SM = W3 + 16777216;              // small pool
  float *Z0=SM, *Z1=Z0+262144, *YH=Z1+262144, *FQ=YH+262144, *FX=FQ+65536,
        *DW=FX+262144, *PW=DW+262144, *CAT=PW+262144, *FEAT=CAT+524288,
        *SEQ=FEAT+262144, *XN=SEQ+262144, *XZ=XN+262144, *XS2=XZ+524288,
        *DBL=XS2+262144, *DT=DBL+540672, *YS=DT+262144, *YG=YS+262144,
        *S2=YG+262144, *MB2=S2+262144, *H1=MB2+262144, *S3=H1+311296,
        *Gb=S3+262144, *BL=Gb+262144, *OC=BL+262144, *AB=OC+262144;
  int* HILp = (int*)(AB + 256);
  int* INVp = HILp + 1024;

  // Hilbert tables
  hil_k<<<4, 256, 0, stream>>>(HILp);
  invhil_k<<<4, 256, 0, stream>>>(HILp, INVp);

  // ---------------- encoder ----------------
  auto dblock = [&](const float* x, int Hin, int Cin, const DBp& p, int stride,
                    float* concat, float* tmp, float* out) {
    long HW = (long)Hin * Hin;
    launch_conv(stream, x, p.b1w, p.b1b, concat, 16, Cin, Hin, Hin, 1,1,1,0, 64, (long)Cin*HW, 128*HW, 0);
    launch_conv(stream, x, p.b2w, p.b2b, tmp,    16, Cin, Hin, Hin, 1,1,1,0, 64, (long)Cin*HW,  64*HW, 0);
    long tot = 16L * 64 * HW;
    dwconv3_k<<<(int)((tot+255)/256), 256, 0, stream>>>(tmp, p.dww, p.dwb, concat, 64, Hin, Hin, 64*HW, 128*HW, 64, tot);
    int Ho = (Hin + 2 - 3) / stride + 1;
    launch_conv(stream, concat, p.fw, p.fb, out, 16, 128, Hin, Hin, 3,3,stride,1, 64, 128*HW, 64L*Ho*Ho, 0);
    inorm_k<<<16 * 64, 256, 0, stream>>>(out, Ho * Ho, 1);
  };
  dblock(input_x, 128, 3,  enc[0], 1, W0, W1, W3);   // enc1 in W3
  dblock(W3,      128, 64, enc[1], 2, W0, W1, W2);
  dblock(W2,       64, 64, enc[2], 1, W0, W1, W1);
  dblock(W1,       64, 64, enc[3], 2, W0, W2, W2);   // embed in W2 (16,64,32,32)

  ri_k<<<1024, 256, 0, stream>>>(W2, ri_w, ri_b, Z0);

  // ---------------- meta blocks ----------------
  auto meta = [&](const MBp& p, float* Zin, float* Zout) {
    haarf_k<<<256, 256, 0, stream>>>(Zin, YH);
    launch_conv(stream, YH, p.fq_w, p.fq_b, FQ, 4, 256, 16,16, 3,3,1,1, 64, 65536L, 16384L, 0);
    bnorm_gelu_k<<<64, 256, 0, stream>>>(FQ, p.fq_g, p.fq_bt, 64, 4, 256);
    upbil_k<<<1024, 256, 0, stream>>>(FQ, FX);
    long tot = 4L * 256 * 256;
    dwconv3_k<<<(int)((tot+255)/256), 256, 0, stream>>>(YH, p.dw_w, p.dw_b, DW, 256, 16, 16, 65536L, 65536L, 0, tot);
    inorm_k<<<4 * 256, 256, 0, stream>>>(DW, 256, 2);
    launch_conv(stream, DW, p.pw_w, p.pw_b, PW, 4, 256, 16,16, 1,1,1,0, 256, 65536L, 65536L, 0);
    bnorm_gelu_k<<<256, 256, 0, stream>>>(PW, p.bn_g, p.bn_b, 256, 4, 256);
    haari_k<<<256, 256, 0, stream>>>(PW, CAT, 131072L, 64);     // wav_out -> CAT[:,64:128]
    // mamba
    add_k<<<1024, 256, 0, stream>>>(Zin, FX, FEAT, 262144);
    hgather_k<<<1024, 256, 0, stream>>>(FEAT, HILp, SEQ);
    lnorm_k<<<64, 256, 0, stream>>>(SEQ, p.ln_g, p.ln_b, XN, 16384, 16);
    lin_k<<<2048, 256, 0, stream>>>(XN, 16, 0, p.in_w, p.in_b, nullptr, 0, XZ, 32, 16384, 16, 32, 0);
    mconv1d_k<<<1024, 256, 0, stream>>>(XZ, p.conv_w, p.conv_b, XS2, 4096);
    lin_k<<<2112, 256, 0, stream>>>(XS2, 16, 0, p.xp_w, nullptr, nullptr, 0, DBL, 33, 16384, 16, 33, 0);
    lin_k<<<1024, 256, 0, stream>>>(DBL, 33, 0, p.dt_w, p.dt_b, nullptr, 0, DT, 16, 16384, 1, 16, 3);
    negexp_k<<<1, 256, 0, stream>>>(p.A_log, AB);
    scan_k<<<4, 256, 0, stream>>>(DT, DBL, XS2, AB, YS, 4096);
    mgate_k<<<1024, 256, 0, stream>>>(YS, XS2, p.D, XZ, YG);
    lin_k<<<1024, 256, 0, stream>>>(YG, 16, 0, p.out_w, p.out_b, SEQ, 16, S2, 16, 16384, 16, 16, 0);
    lnorm_k<<<64, 256, 0, stream>>>(S2, p.ln2_g, p.ln2_b, MB2, 16384, 16);
    lin_k<<<1216, 256, 0, stream>>>(MB2, 16, 0, p.w1, p.b1, nullptr, 0, H1, 19, 16384, 16, 19, 2);
    lin_k<<<1024, 256, 0, stream>>>(H1, 19, 0, p.w2, p.b2, S2, 16, S3, 16, 16384, 19, 16, 0);
    hscatter_k<<<1024, 256, 0, stream>>>(S3, INVp, CAT);        // m -> CAT[:,0:64]
    // gate + output
    launch_conv(stream, CAT, p.g_w, p.g_b, Gb, 4, 128, 32,32, 3,3,1,1, 64, 131072L, 65536L, 0);
    blend_k<<<1024, 256, 0, stream>>>(Gb, CAT, BL);
    launch_conv(stream, BL, p.o_w, p.o_b, OC, 4, 64, 32,32, 1,1,1,0, 64, 65536L, 65536L, 0);
    add_k<<<1024, 256, 0, stream>>>(OC, Zin, Zout, 262144);
  };
  meta(hid[0], Z0, Z1);
  meta(hid[1], Z1, Z0);
  meta(hid[2], Z0, Z1);

  ro_k<<<4096, 256, 0, stream>>>(Z1, ro_w, ro_b, W2);           // fc (16,64,32,32) in W2

  // ---------------- decoder ----------------
  launch_conv(stream, W2, dec[0].w, dec[0].b, W1, 16, 64, 32, 32, 3,3,1,1, 256, 64L*1024, 256L*1024, 0);
  { long t = 16L*256*1024; pshuf_k<<<(int)((t+255)/256), 256, 0, stream>>>(W1, W0, 16, 256, 32, 32); }
  launch_conv(stream, W0, dec[1].w, dec[1].b, W1, 16, 64, 64, 64, 3,3,1,1, 64, 64L*4096, 64L*4096, 0);
  launch_conv(stream, W1, dec[2].w, dec[2].b, W0, 16, 64, 64, 64, 3,3,1,1, 256, 64L*4096, 256L*4096, 0);
  { long t = 16L*256*4096; pshuf_k<<<(int)((t+255)/256), 256, 0, stream>>>(W0, W1, 16, 256, 64, 64); }
  add_k<<<65536, 256, 0, stream>>>(W1, W3, W1, 16777216);       // + enc1
  launch_conv(stream, W1, dec[3].w, dec[3].b, W0, 16, 64, 128, 128, 3,3,1,1, 64, 64L*16384, 64L*16384, 0);
  launch_conv(stream, W0, dec[4].w, dec[4].b, W1, 16, 64, 128, 128, 1,1,1,0, 3, 64L*16384, 3L*16384, 0);

  // ---------------- clip + loss ----------------
  hipMemsetAsync((float*)d_out + 786432, 0, sizeof(float), stream);
  finalize_k<<<768, 256, 0, stream>>>(W1, targets, (float*)d_out, (float*)d_out + 786432, 786432);
}
